// RSAGEIIConv_6150393168695
// MI455X (gfx1250) — compile-verified, matrix operand order checked by reading, not run on hardware
//
#include <hip/hip_runtime.h>

#define BB 2
#define CC 128
#define NN 20000
#define KK 16

typedef __attribute__((ext_vector_type(16))) __bf16 v16bf;
typedef __attribute__((ext_vector_type(8)))  float  v8f;

__device__ __forceinline__ unsigned short f2bfu(float f) {
    unsigned u = __builtin_bit_cast(unsigned, f);
    u += 0x7fffu + ((u >> 16) & 1u);           // round-to-nearest-even
    return (unsigned short)(u >> 16);
}

// Load a 16-half fragment as two aligned 16-byte chunks (-> 2x b128 loads).
__device__ __forceinline__ v16bf load16(const unsigned short* p0,
                                        const unsigned short* p1) {
    union { uint4 q[2]; v16bf v; } u;
    u.q[0] = *(const uint4*)p0;
    u.q[1] = *(const uint4*)p1;
    return u.v;
}

// Fragment from a row-major [*, ld] bf16 source: lane (half,row) needs halves
// K = {kbase+half*8 .. +7} and {kbase+16+half*8 .. +7} -> two contiguous chunks.
__device__ __forceinline__ v16bf load_frag(const unsigned short* base, int row,
                                           int kbase, int half, int ld) {
    const unsigned short* p = base + row * ld + kbase + half * 8;
    return load16(p, p + 16);
}

// ---------------------------------------------------------------------------
// Stage A: pack all four weights in WMMA B-fragment order
// [o_tile(8)][k_step(4)][lane(32)][elem(16)], element (K=c, N=o) = w[o][c].
// w1, w2 are pre-scaled by BETA. Also zero the BN stat accumulators.
// ---------------------------------------------------------------------------
__global__ __launch_bounds__(256) void k_prep(
    const float* __restrict__ wpre, const float* __restrict__ w1,
    const float* __restrict__ w2, const float* __restrict__ wr,
    unsigned short* __restrict__ wprep, unsigned short* __restrict__ w1p,
    unsigned short* __restrict__ w2p, unsigned short* __restrict__ wrp,
    float* __restrict__ stats) {
    const int i = blockIdx.x * 256 + threadIdx.x;
    if (i < CC * CC) {
        const float BETA = 0.40546510810816438f;
        const int j    = i & 15;
        const int lane = (i >> 4) & 31;
        const int ks   = (i >> 9) & 3;
        const int ot   = i >> 11;
        const int half = lane >> 4, col = lane & 15;
        const int o  = ot * 16 + col;
        const int jj = (j < 8) ? j : j + 8;       // skip to K+16 for upper half
        const int c  = ks * 32 + half * 8 + jj;
        wprep[i] = f2bfu(wpre[o * CC + c]);
        w1p[i]   = f2bfu(BETA * w1[o * CC + c]);
        w2p[i]   = f2bfu(BETA * w2[o * CC + c]);
        wrp[i]   = f2bfu(wr[o * CC + c]);
    }
    if (blockIdx.x == 0 && threadIdx.x < 2 * CC) stats[threadIdx.x] = 0.f;
}

// ---------------------------------------------------------------------------
// Stage B: gather neighbors -> bf16 LDS [k][c]; compute Y^T = X_j^T @ W_pre^T
// (M = 16 neighbors, N = 16 out-channels, K = 128 channels): max over
// neighbors is an in-register v_max3 reduction + one shfl_xor(16).
// B-fragments are double-buffered one o-tile ahead so global loads overlap
// the WMMAs. One wave per node, per-wave LDS region, no block barrier.
// ---------------------------------------------------------------------------
__global__ __launch_bounds__(256) void k_gather_prenn_max(
    const float* __restrict__ x, const long long* __restrict__ eidx,
    const unsigned short* __restrict__ wprep, float* __restrict__ aggr) {
    __shared__ unsigned short glds[8][KK * CC];  // 32 KB: per-wave [k][c] bf16

    const int tid = threadIdx.x;
    const int wave = tid >> 5, lane = tid & 31;
    const int g = blockIdx.x * 8 + wave;         // node id in [0, B*N)
    const int b = g / NN, n = g % NN;

    // neighbor indices: lane k (and 16+k) loads index k, broadcast via shfl
    const long long ebase = ((long long)b * NN + n) * KK;
    const int jl = (int)eidx[ebase + (lane & 15)];

    // gather 16 rows x_flat[b,:,j] -> LDS [k][c]; pack 2 bf16 per b32 store
#pragma unroll 4
    for (int k = 0; k < KK; ++k) {
        const int j = __shfl(jl, k, 32);
#pragma unroll
        for (int h = 0; h < 2; ++h) {
            const int c = 2 * lane + 64 * h;
            const unsigned lo = f2bfu(x[((size_t)b * CC + c) * NN + j]);
            const unsigned hi = f2bfu(x[((size_t)b * CC + c + 1) * NN + j]);
            *(unsigned*)&glds[wave][k * CC + c] = lo | (hi << 16);
        }
    }
    __builtin_amdgcn_wave_barrier();             // keep gather before frag loads

    const int half = lane >> 4, col = lane & 15;
    // A-fragments (neighbors x channels) are o-tile invariant: load once.
    v16bf afr[4];
#pragma unroll
    for (int ks = 0; ks < 4; ++ks)
        afr[ks] = load_frag(glds[wave], col, ks * 32, half, CC);

    const unsigned short* wl = wprep + lane * 16;  // per-lane packed base
    // double-buffered weight B-fragments (one o-tile = 4 fragments ahead)
    v16bf bw[4], bwn[4];
#pragma unroll
    for (int ks = 0; ks < 4; ++ks)
        bw[ks] = load16(wl + ks * 512, wl + ks * 512 + 8);

#pragma unroll
    for (int ot = 0; ot < 8; ++ot) {             // 8 tiles of 16 out-channels
        if (ot < 7) {
            const unsigned short* wn = wl + (ot + 1) * 2048;
#pragma unroll
            for (int ks = 0; ks < 4; ++ks)
                bwn[ks] = load16(wn + ks * 512, wn + ks * 512 + 8);
        }
        v8f acc = {};
#pragma unroll
        for (int ks = 0; ks < 4; ++ks)
            acc = __builtin_amdgcn_wmma_f32_16x16x32_bf16(
                false, afr[ks], false, bw[ks], (short)0, acc, false, false);
        // relu + max over M (neighbors): 8 VGPRs in-lane, then other lane-half
        float m = fmaxf(acc[0], 0.f);
#pragma unroll
        for (int i = 1; i < 8; ++i) m = fmaxf(m, acc[i]);
        m = fmaxf(m, __shfl_xor(m, 16, 32));
        if (lane < 16) aggr[(size_t)g * CC + ot * 16 + lane] = m;
#pragma unroll
        for (int ks = 0; ks < 4; ++ks) bw[ks] = bwn[ks];
    }
}

// ---------------------------------------------------------------------------
// Stage C: out = relu( c1*aggr + c2*x0 + aggr@(B*w1^T) + x0@(B*w2^T) + x^T@w_r^T )
// One 16-node tile per block; each wave owns one 16-channel o-tile (12 WMMAs).
// Weight B-fragments are double-buffered one k-step ahead.
// ---------------------------------------------------------------------------
__global__ __launch_bounds__(256) void k_fuse_linear(
    const float* __restrict__ x, const float* __restrict__ x0,
    const float* __restrict__ aggr,
    const unsigned short* __restrict__ w1p, const unsigned short* __restrict__ w2p,
    const unsigned short* __restrict__ wrp, float* __restrict__ mid) {
    __shared__ unsigned short a_ag[16 * CC], a_x0[16 * CC], a_xr[16 * CC]; // 12 KB

    const int tid = threadIdx.x;
    const int t = blockIdx.x;                    // node-tile id
    const int b = t / (NN / 16), n0 = (t % (NN / 16)) * 16;

    for (int i = tid; i < 16 * CC; i += 256) {
        const int row = i >> 7, c = i & 127;
        const size_t g = (size_t)b * NN + n0 + row;
        a_ag[i] = f2bfu(aggr[g * CC + c]);
        a_x0[i] = f2bfu(x0[g * CC + c]);
        a_xr[i] = f2bfu(x[((size_t)b * CC + c) * NN + (n0 + row)]);
    }
    __syncthreads();

    const int wave = tid >> 5, lane = tid & 31;
    const int half = lane >> 4, col = lane & 15;
    const int oc = wave * 16 + col;              // output channel

    const int pbase = (wave * 4 * 32 + lane) * 16;  // packed base for this wave
    v16bf b1, b2, br, b1n, b2n, brn;
    b1 = load16(w1p + pbase, w1p + pbase + 8);
    b2 = load16(w2p + pbase, w2p + pbase + 8);
    br = load16(wrp + pbase, wrp + pbase + 8);

    v8f acc = {};
#pragma unroll
    for (int ks = 0; ks < 4; ++ks) {
        if (ks < 3) {
            const int pn = pbase + (ks + 1) * 512;
            b1n = load16(w1p + pn, w1p + pn + 8);
            b2n = load16(w2p + pn, w2p + pn + 8);
            brn = load16(wrp + pn, wrp + pn + 8);
        }
        v16bf aa = load_frag(a_ag, col, ks * 32, half, CC);
        acc = __builtin_amdgcn_wmma_f32_16x16x32_bf16(false, aa, false, b1, (short)0, acc, false, false);
        v16bf a0 = load_frag(a_x0, col, ks * 32, half, CC);
        acc = __builtin_amdgcn_wmma_f32_16x16x32_bf16(false, a0, false, b2, (short)0, acc, false, false);
        v16bf ar = load_frag(a_xr, col, ks * 32, half, CC);
        acc = __builtin_amdgcn_wmma_f32_16x16x32_bf16(false, ar, false, br, (short)0, acc, false, false);
        b1 = b1n; b2 = b2n; br = brn;
    }

    const float BETA = 0.40546510810816438f;
    const float c1 = 0.9f * (1.0f - BETA);       // (1-ALPHA)*(1-BETA)
    const float c2 = 0.1f * (1.0f - BETA);       // ALPHA*(1-BETA)
#pragma unroll
    for (int r = 0; r < 8; ++r) {
        const int M = half * 8 + r;              // node row within tile
        const size_t g = (size_t)b * NN + n0 + M;
        const size_t idx = g * CC + oc;
        const float v = acc[r] + c1 * aggr[idx] + c2 * x0[idx];
        mid[idx] = fmaxf(v, 0.f);
    }
}

// ---------------------------------------------------------------------------
// Stage D: per-channel sum / sumsq partials (coalesced rows, atomic combine).
// ---------------------------------------------------------------------------
__global__ __launch_bounds__(128) void k_stats(const float* __restrict__ mid,
                                               float* __restrict__ stats) {
    const int c = threadIdx.x;                   // 128 channels
    const int rows_per = (BB * NN) / gridDim.x;
    const int r0 = blockIdx.x * rows_per;
    float s = 0.f, sq = 0.f;
#pragma unroll 4
    for (int r = 0; r < rows_per; ++r) {
        const float v = mid[(size_t)(r0 + r) * CC + c];
        s += v; sq += v * v;
    }
    atomicAdd(&stats[c], s);
    atomicAdd(&stats[CC + c], sq);
}

// ---------------------------------------------------------------------------
// Stage E: fold mean/var/gamma/beta/bias into per-channel scale + shift.
// ---------------------------------------------------------------------------
__global__ void k_finalize_stats(const float* __restrict__ stats,
                                 const float* __restrict__ gamma,
                                 const float* __restrict__ beta,
                                 const float* __restrict__ bias,
                                 float* __restrict__ ss) {
    const int c = threadIdx.x;
    if (c < CC) {
        const float inv_n = 1.0f / (float)(BB * NN);
        const float mean = stats[c] * inv_n;
        const float var  = stats[CC + c] * inv_n - mean * mean;
        const float is   = rsqrtf(var + 1e-5f);
        const float sc   = gamma[c] * is;
        ss[c]      = sc;
        ss[CC + c] = beta[c] - mean * sc + bias[c];
    }
}

// ---------------------------------------------------------------------------
// Stage F: BN apply + L2-normalize over channels, transposed store [B,C,N,1].
// One wave per node, 4 channels per lane, shfl_xor norm reduction.
// ---------------------------------------------------------------------------
__global__ __launch_bounds__(256) void k_norm_out(const float* __restrict__ mid,
                                                  const float* __restrict__ ss,
                                                  float* __restrict__ out) {
    const int wave = threadIdx.x >> 5, lane = threadIdx.x & 31;
    const int g = blockIdx.x * 8 + wave;
    const int b = g / NN, n = g % NN;
    float v[4];
    float s = 0.f;
#pragma unroll
    for (int j = 0; j < 4; ++j) {
        const int c = lane + 32 * j;
        v[j] = mid[(size_t)g * CC + c] * ss[c] + ss[CC + c];
        s += v[j] * v[j];
    }
#pragma unroll
    for (int m = 16; m >= 1; m >>= 1) s += __shfl_xor(s, m, 32);
    const float inv = 1.0f / fmaxf(sqrtf(s), 1e-12f);
#pragma unroll
    for (int j = 0; j < 4; ++j) {
        const int c = lane + 32 * j;
        out[((size_t)b * CC + c) * NN + n] = v[j] * inv;
    }
}

extern "C" void kernel_launch(void* const* d_in, const int* in_sizes, int n_in,
                              void* d_out, int out_size, void* d_ws, size_t ws_size,
                              hipStream_t stream) {
    (void)in_sizes; (void)n_in; (void)out_size; (void)ws_size;
    const float*     x    = (const float*)d_in[0];
    const float*     x0   = (const float*)d_in[1];
    const long long* e    = (const long long*)d_in[2];   // int64 edge_index
    const float*     wpre = (const float*)d_in[3];
    const float*     w1   = (const float*)d_in[4];
    const float*     w2   = (const float*)d_in[5];
    const float*     wr   = (const float*)d_in[6];
    const float*     gam  = (const float*)d_in[7];
    const float*     bet  = (const float*)d_in[8];
    const float*     bias = (const float*)d_in[9];

    char* ws = (char*)d_ws;
    unsigned short* wprep = (unsigned short*)(ws + 0);
    unsigned short* w1p   = (unsigned short*)(ws + 32768);
    unsigned short* w2p   = (unsigned short*)(ws + 65536);
    unsigned short* wrp   = (unsigned short*)(ws + 98304);
    float* stats          = (float*)(ws + 131072);
    float* ss             = (float*)(ws + 132096);
    float* aggr           = (float*)(ws + 135168);
    float* mid            = (float*)(ws + 135168 + (size_t)BB * NN * CC * 4);
    float* out = (float*)d_out;

    k_prep<<<64, 256, 0, stream>>>(wpre, w1, w2, wr, wprep, w1p, w2p, wrp, stats);
    k_gather_prenn_max<<<(BB * NN) / 8, 256, 0, stream>>>(x, e, wprep, aggr);
    k_fuse_linear<<<BB * (NN / 16), 256, 0, stream>>>(x, x0, aggr, w1p, w2p, wrp, mid);
    k_stats<<<250, 128, 0, stream>>>(mid, stats);
    k_finalize_stats<<<1, 128, 0, stream>>>(stats, gam, bet, bias, ss);
    k_norm_out<<<(BB * NN) / 8, 256, 0, stream>>>(mid, ss, out);
}
